// MHA2_32066225832319
// MI455X (gfx1250) — matrix operand order repaired and checked by reading, not compiled, not run on hardware
//
#include <hip/hip_runtime.h>
#include <hip/hip_bf16.h>

#define B_ 2
#define S_ 2048
#define D_ 2048
#define H_ 16
#define DK_ 128
#define SCALE_ 0.08838834764831845f  // 1/sqrt(128)
#define LS 56     // stride for 32-wide staging tiles (k_proj)
#define LQ 136    // stride for resident 128-wide Q tile / K strip
#define LP 40     // stride for 32-wide p/V tiles

typedef __bf16 bf16_t;
typedef bf16_t v16bf __attribute__((ext_vector_type(16)));
typedef bf16_t v8bf  __attribute__((ext_vector_type(8)));
typedef bf16_t v4bf  __attribute__((ext_vector_type(4)));
typedef float  v8f   __attribute__((ext_vector_type(8)));

__device__ __forceinline__ v16bf mk16(v8bf lo, v8bf hi) {
  return __builtin_shufflevector(lo, hi, 0,1,2,3,4,5,6,7,8,9,10,11,12,13,14,15);
}

// A 16x32 bf16 fragment: lane<16 -> K 0..7 / 16..23 ; lane>=16 -> K 8..15 / 24..31
__device__ __forceinline__ v16bf load_a_frag(const bf16_t* lds, int stride, int kcOff,
                                             int lane, int w) {
  const int rowA = w * 16 + (lane & 15);
  const int h0 = (lane >> 4) * 8;
  const bf16_t* p = lds + rowA * stride + kcOff;
  v8bf lo = *(const v8bf*)(p + h0);
  v8bf hi = *(const v8bf*)(p + h0 + 16);
  return mk16(lo, hi);
}

// B 32x16 bf16 fragment: N = lane%16, lane<16 -> K 0..7 / 16..23 ; lane>=16 -> K 8..15 / 24..31
__device__ __forceinline__ v16bf load_b_frag(const bf16_t* lds, int stride, int kcOff,
                                             int lane, int t) {
  const bf16_t* p = lds + (t * 16 + (lane & 15)) * stride + kcOff + ((lane >> 4) * 8);
  v8bf lo = *(const v8bf*)p;
  v8bf hi = *(const v8bf*)(p + 16);
  return mk16(lo, hi);
}

// ---------------------------------------------------------------------------
// Kernel 1 (flash-style attention, two in-block passes):
// grid = (qtiles=16, B*H=32), block = 256 (8 waves).
// Q tile (128x128 bf16) is LDS-resident for the whole kernel.
// Pass A: loop 64 key strips (32 wide), 8 WMMA each, online per-lane
//         (running max, running sum-exp); xor-shuffle merge -> exact row stats.
// Pass B: loop key strips again: recompute scores, normalize, write f32
//         p_attn strip (the one mandatory big write), bounce bf16 p through
//         LDS, X += p * V via WMMA.
// LDS: Q 34816B + K 8704B + V 10240B + P 10240B = 64000B.
// ---------------------------------------------------------------------------
__global__ __launch_bounds__(256) void k_attn(const float* __restrict__ Q,
                                              const float* __restrict__ K,
                                              const float* __restrict__ V,
                                              float* __restrict__ outP,
                                              bf16_t* __restrict__ X) {
  __shared__ bf16_t ldsQ[128 * LQ];  // resident Q tile
  __shared__ bf16_t ldsK[32 * LQ];   // K strip (32 keys x 128 feat)
  __shared__ bf16_t ldsV[128 * LP];  // V strip transposed [d][k]
  __shared__ bf16_t ldsP[128 * LP];  // normalized p strip [q][k]
  const int tid = threadIdx.x;
  const int lane = tid & 31, w = tid >> 5;
  const int qt = blockIdx.x, bh = blockIdx.y;
  const int b = bh >> 4, h = bh & 15;
  const int q0 = qt * 128;
  const size_t qbase = ((size_t)b * S_ + q0) * D_ + (size_t)h * DK_;
  float* prow = outP + (size_t)bh * S_ * S_;
  const int rloc = w * 16 + ((lane >> 4) << 3);  // local row base for this lane

  // ---- stage resident Q tile (f32 -> bf16), 16 float4 per thread ----
#pragma unroll
  for (int cp = 0; cp < 4; ++cp) {
#pragma unroll
    for (int rp = 0; rp < 4; ++rp) {
      const int r = rp * 32 + (tid >> 3);
      const int c = cp * 32 + (tid & 7) * 4;
      float4 qv = *(const float4*)(Q + qbase + (size_t)r * D_ + c);
      v4bf qa = {(bf16_t)qv.x, (bf16_t)qv.y, (bf16_t)qv.z, (bf16_t)qv.w};
      *(v4bf*)&ldsQ[r * LQ + c] = qa;
    }
  }

  v8f zero = {};

  // =============== Pass A: online softmax statistics ===============
  float om[8], os[8];
#pragma unroll
  for (int r = 0; r < 8; ++r) { om[r] = -1e30f; os[r] = 0.f; }

  for (int kt = 0; kt < S_ / 32; ++kt) {
    const size_t kbase = ((size_t)b * S_ + kt * 32) * D_ + (size_t)h * DK_;
    __syncthreads();
#pragma unroll
    for (int cp = 0; cp < 4; ++cp) {
      const int r = tid >> 3;               // 0..31 (key)
      const int c = cp * 32 + (tid & 7) * 4;
      float4 kv = *(const float4*)(K + kbase + (size_t)r * D_ + c);
      v4bf ka = {(bf16_t)kv.x, (bf16_t)kv.y, (bf16_t)kv.z, (bf16_t)kv.w};
      *(v4bf*)&ldsK[r * LQ + c] = ka;
    }
    __syncthreads();

    v8f S0 = zero, S1 = zero;
#pragma unroll
    for (int kc = 0; kc < 4; ++kc) {
      v16bf A = load_a_frag(ldsQ, LQ, kc * 32, lane, w);
      v16bf B0 = load_b_frag(ldsK, LQ, kc * 32, lane, 0);
      v16bf B1 = load_b_frag(ldsK, LQ, kc * 32, lane, 1);
      S0 = __builtin_amdgcn_wmma_f32_16x16x32_bf16(false, A, false, B0, (short)0, S0, false, false);
      S1 = __builtin_amdgcn_wmma_f32_16x16x32_bf16(false, A, false, B1, (short)0, S1, false, false);
    }
#pragma unroll
    for (int r = 0; r < 8; ++r) {
      const float v0 = S0[r] * SCALE_, v1 = S1[r] * SCALE_;
      const float mn = fmaxf(om[r], fmaxf(v0, v1));
      os[r] = os[r] * expf(om[r] - mn) + expf(v0 - mn) + expf(v1 - mn);
      om[r] = mn;
    }
  }

  // cross-lane merge (16 lanes share each row); exact softmax stats
  float mr[8], ir[8];
#pragma unroll
  for (int r = 0; r < 8; ++r) {
#pragma unroll
    for (int off = 1; off < 16; off <<= 1) {
      const float m2 = __shfl_xor(om[r], off, 32);
      const float s2 = __shfl_xor(os[r], off, 32);
      const float mn = fmaxf(om[r], m2);
      os[r] = os[r] * expf(om[r] - mn) + s2 * expf(m2 - mn);
      om[r] = mn;
    }
    mr[r] = om[r];
    ir[r] = 1.0f / os[r];
  }

  // =============== Pass B: p_attn write + X = p * V ===============
  v8f Xacc[8];
#pragma unroll
  for (int i = 0; i < 8; ++i) Xacc[i] = zero;

  for (int kt = 0; kt < S_ / 32; ++kt) {
    const int n0 = kt * 32;
    const size_t kbase = ((size_t)b * S_ + n0) * D_ + (size_t)h * DK_;

    __syncthreads();
#pragma unroll
    for (int cp = 0; cp < 4; ++cp) {
      const int r = tid >> 3;
      const int c = cp * 32 + (tid & 7) * 4;
      float4 kv = *(const float4*)(K + kbase + (size_t)r * D_ + c);
      v4bf ka = {(bf16_t)kv.x, (bf16_t)kv.y, (bf16_t)kv.z, (bf16_t)kv.w};
      *(v4bf*)&ldsK[r * LQ + c] = ka;
    }
#pragma unroll
    for (int pass = 0; pass < 4; ++pass) {  // V strip -> ldsV[d][k]
      const int k = pass * 8 + (tid >> 5);
      const int n4 = (tid & 31) * 4;
      float4 vv = *(const float4*)(V + ((size_t)b * S_ + n0 + k) * D_ +
                                   (size_t)h * DK_ + n4);
      ldsV[(n4 + 0) * LP + k] = (bf16_t)vv.x;
      ldsV[(n4 + 1) * LP + k] = (bf16_t)vv.y;
      ldsV[(n4 + 2) * LP + k] = (bf16_t)vv.z;
      ldsV[(n4 + 3) * LP + k] = (bf16_t)vv.w;
    }
    __syncthreads();

    v8f S0 = zero, S1 = zero;
#pragma unroll
    for (int kc = 0; kc < 4; ++kc) {
      v16bf A = load_a_frag(ldsQ, LQ, kc * 32, lane, w);
      v16bf B0 = load_b_frag(ldsK, LQ, kc * 32, lane, 0);
      v16bf B1 = load_b_frag(ldsK, LQ, kc * 32, lane, 1);
      S0 = __builtin_amdgcn_wmma_f32_16x16x32_bf16(false, A, false, B0, (short)0, S0, false, false);
      S1 = __builtin_amdgcn_wmma_f32_16x16x32_bf16(false, A, false, B1, (short)0, S1, false, false);
    }

    // normalize + write p_attn strip (f32, coalesced)
#pragma unroll
    for (int r = 0; r < 8; ++r) {
      S0[r] = expf(S0[r] * SCALE_ - mr[r]) * ir[r];
      S1[r] = expf(S1[r] * SCALE_ - mr[r]) * ir[r];
    }
    {
      const int c0 = n0 + (lane & 15);
#pragma unroll
      for (int r = 0; r < 8; ++r) {
        float* pr = prow + (size_t)(q0 + rloc + r) * S_;
        pr[c0] = S0[r];
        pr[c0 + 16] = S1[r];
      }
    }

    // bounce bf16 p through LDS, then X += p * V (8 WMMA)
    {
      const int cl = lane & 15;
#pragma unroll
      for (int r = 0; r < 8; ++r) {
        ldsP[(rloc + r) * LP + cl] = (bf16_t)S0[r];
        ldsP[(rloc + r) * LP + cl + 16] = (bf16_t)S1[r];
      }
    }
    __syncthreads();
    v16bf A = load_a_frag(ldsP, LP, 0, lane, w);
#pragma unroll
    for (int t = 0; t < 8; ++t) {
      v16bf Bf = load_b_frag(ldsV, LP, 0, lane, t);
      Xacc[t] = __builtin_amdgcn_wmma_f32_16x16x32_bf16(
          false, A, false, Bf, (short)0, Xacc[t], false, false);
    }
  }

  // ---- write X (bf16, combined-head [B,S,D] layout) ----
#pragma unroll
  for (int t = 0; t < 8; ++t) {
    const int col = t * 16 + (lane & 15);
#pragma unroll
    for (int r = 0; r < 8; ++r)
      X[((size_t)b * S_ + q0 + rloc + r) * D_ + (size_t)h * DK_ + col] =
          (bf16_t)Xacc[t][r];
  }
}

// ---------------------------------------------------------------------------
// Kernel 2: out[m,o] = sum_i X[m,i] * Wo[o,i] + bias[o]   (M = B*S = 4096)
// grid = (ntiles=16, mtiles=32).
// ---------------------------------------------------------------------------
__global__ __launch_bounds__(256) void k_proj(const bf16_t* __restrict__ X,
                                              const float* __restrict__ Wo,
                                              const float* __restrict__ bias,
                                              float* __restrict__ out) {
  __shared__ bf16_t ldsA[128 * LS];
  __shared__ bf16_t ldsB[128 * LS];
  const int tid = threadIdx.x;
  const int lane = tid & 31, w = tid >> 5;
  const int m0 = blockIdx.y * 128, n0 = blockIdx.x * 128;

  v8f zero = {};
  v8f acc[8];
#pragma unroll
  for (int i = 0; i < 8; ++i) acc[i] = zero;

  for (int kb = 0; kb < D_; kb += 32) {
#pragma unroll
    for (int pass = 0; pass < 4; ++pass) {
      const int r = pass * 32 + (tid >> 3);
      const int c = (tid & 7) * 4;
      v4bf xa = *(const v4bf*)(X + (size_t)(m0 + r) * D_ + kb + c);
      *(v4bf*)&ldsA[r * LS + c] = xa;
      float4 wv = *(const float4*)(Wo + (size_t)(n0 + r) * D_ + kb + c);
      v4bf wa = {(bf16_t)wv.x, (bf16_t)wv.y, (bf16_t)wv.z, (bf16_t)wv.w};
      *(v4bf*)&ldsB[r * LS + c] = wa;
    }
    __syncthreads();
    v16bf A = load_a_frag(ldsA, LS, 0, lane, w);
#pragma unroll
    for (int t = 0; t < 8; ++t) {
      v16bf Bf = load_b_frag(ldsB, LS, 0, lane, t);
      acc[t] = __builtin_amdgcn_wmma_f32_16x16x32_bf16(
          false, A, false, Bf, (short)0, acc[t], false, false);
    }
    __syncthreads();
  }

  const int rbase = m0 + w * 16 + ((lane >> 4) << 3);
#pragma unroll
  for (int t = 0; t < 8; ++t) {
    const int col = n0 + t * 16 + (lane & 15);
    const float bb = bias[col];
#pragma unroll
    for (int r = 0; r < 8; ++r)
      out[(size_t)(rbase + r) * D_ + col] = acc[t][r] + bb;
  }
}

// ---------------------------------------------------------------------------
extern "C" void kernel_launch(void* const* d_in, const int* in_sizes, int n_in,
                              void* d_out, int out_size, void* d_ws, size_t ws_size,
                              hipStream_t stream) {
  (void)in_sizes; (void)n_in; (void)out_size; (void)ws_size;
  const float* Q  = (const float*)d_in[0];
  const float* K  = (const float*)d_in[1];
  const float* V  = (const float*)d_in[2];
  const float* Wo = (const float*)d_in[3];
  const float* Wb = (const float*)d_in[4];

  float* outO = (float*)d_out;
  float* outP = outO + (size_t)B_ * S_ * D_;  // p_attn region [B,H,S,S]

  bf16_t* Xb = (bf16_t*)d_ws;  // 16 MB bf16 [B,S,D]

  dim3 blk(256);
  k_attn<<<dim3(S_ / 128, B_ * H_), blk, 0, stream>>>(Q, K, V, outP, Xb);
  k_proj<<<dim3(D_ / 128, (B_ * S_) / 128), blk, 0, stream>>>(Xb, Wo, Wb, outO);
}
